// CausalSelfAttention_5592047420110
// MI455X (gfx1250) — matrix operand order repaired and checked by reading, not compiled, not run on hardware
//
#include <hip/hip_runtime.h>
#include <hip/hip_bf16.h>

typedef __attribute__((ext_vector_type(16))) __bf16 v16bf;
typedef __attribute__((ext_vector_type(8)))  __bf16 v8bf;
typedef __attribute__((ext_vector_type(4)))  __bf16 v4bf;
typedef __attribute__((ext_vector_type(8)))  float  v8f;

#define T_SEQ 2048
#define C_DIM 1024
#define N_HEAD 16
#define HEAD_DIM 64
#define ROWS 8192            // B*T
#define LN10000 9.210340371976184f

#define KSTEP 64             // K per LDS stage (2 WMMA k-chunks)
#define LDSROW 72            // 64 + 8 pad elements -> 144B row, conflict-free B128

union BF16x16 { v16bf v; v8bf h[2]; };

// ---- fragment loaders (layouts per CDNA5 ISA 7.12.2) -----------------------
// A 16x32 bf16: lane l -> row m=l&15; elems 0..7 = K[hi*8..], elems 8..15 = K[16+hi*8..]
__device__ __forceinline__ v16bf a_frag(const __bf16* rowp, int k0, int hi) {
    BF16x16 u;
    u.h[0] = *(const v8bf*)(rowp + k0 + hi * 8);
    u.h[1] = *(const v8bf*)(rowp + k0 + 16 + hi * 8);
    return u.v;
}
// B 32x16 bf16: lane l -> col n=l&15; elems 0..7 = K[hi*8..], elems 8..15 = K[16+hi*8..]
__device__ __forceinline__ v16bf b_frag(const __bf16* colp, int k0, int hi) {
    BF16x16 u;
    u.h[0] = *(const v8bf*)(colp + k0 + hi * 8);
    u.h[1] = *(const v8bf*)(colp + k0 + 16 + hi * 8);
    return u.v;
}
__device__ __forceinline__ v8f wmma_bf16(v16bf a, v16bf b, v8f c) {
    return __builtin_amdgcn_wmma_f32_16x16x32_bf16(false, a, false, b, (short)0, c, false, false);
}

// ---- gfx1250 async Global->LDS copy (ASYNCcnt path) ------------------------
__device__ __forceinline__ void async_cp16(unsigned lds_off, const __bf16* g) {
    asm volatile("global_load_async_to_lds_b128 %0, %1, off"
                 :: "v"(lds_off), "v"((unsigned long long)(uintptr_t)g)
                 : "memory");
}
#define WAIT_ASYNC(n) asm volatile("s_wait_asynccnt " #n ::: "memory")

// Stage one K-step of tiles: A 128x64 (4 chunks/thread), B 64x64 (2 chunks/thread)
__device__ __forceinline__ void stage_tiles(const __bf16* __restrict__ gA,
                                            const __bf16* __restrict__ gB,
                                            unsigned ldsA, unsigned ldsB,
                                            int tid, int k0) {
#pragma unroll
    for (int i = 0; i < 4; ++i) {                 // 1024 chunks of 16B
        int chunk = tid + i * 256;
        int row = chunk >> 3, cc = chunk & 7;
        async_cp16(ldsA + row * (LDSROW * 2) + cc * 16, gA + row * C_DIM + k0 + cc * 8);
    }
#pragma unroll
    for (int i = 0; i < 2; ++i) {                 // 512 chunks of 16B
        int chunk = tid + i * 256;
        int row = chunk >> 3, cc = chunk & 7;
        async_cp16(ldsB + row * (LDSROW * 2) + cc * 16, gB + row * C_DIM + k0 + cc * 8);
    }
}

// ---- f32 -> bf16 convert ---------------------------------------------------
__global__ void cvt_f32_bf16(const float* __restrict__ in, __bf16* __restrict__ out, int n) {
    int i = (blockIdx.x * blockDim.x + threadIdx.x) * 4;
    if (i < n) {
        float4 f = *(const float4*)(in + i);
        v4bf o;
        o.x = (__bf16)f.x; o.y = (__bf16)f.y; o.z = (__bf16)f.z; o.w = (__bf16)f.w;
        *(v4bf*)(out + i) = o;
    }
}

// ---- QKV GEMM + bias + RoPE epilogue (LDS-staged, double buffered) --------
// grid: (ROWS/128, C/64, 3)  block: 256 (8 waves; 4 M-waves x 2 N-waves)
__global__ __launch_bounds__(256)
void qkv_gemm_rope(const __bf16* __restrict__ xb,
                   const __bf16* __restrict__ wq, const __bf16* __restrict__ wk,
                   const __bf16* __restrict__ wv,
                   const float* __restrict__ bq, const float* __restrict__ bk,
                   const float* __restrict__ bv,
                   __bf16* __restrict__ qws, __bf16* __restrict__ kws,
                   __bf16* __restrict__ vtws,
                   float* __restrict__ kout, float* __restrict__ vout) {
    __shared__ __align__(16) __bf16 As[2][128][LDSROW];   // 36864 B
    __shared__ __align__(16) __bf16 Bs[2][64][LDSROW];    // 18432 B

    const int z = blockIdx.z;
    const __bf16* W    = (z == 0) ? wq : (z == 1) ? wk : wv;
    const float*  bias = (z == 0) ? bq : (z == 1) ? bk : bv;

    const int tid  = threadIdx.x;
    const int lane = tid & 31;
    const int wid  = tid >> 5;
    const int wm = wid & 3, wn = wid >> 2;
    const int Rblk = blockIdx.x * 128;
    const int Cblk = blockIdx.y * 64;
    const int ln = lane & 15, hi = lane >> 4;

    const __bf16* gA = xb + Rblk * C_DIM;
    const __bf16* gB = W  + Cblk * C_DIM;
    const unsigned ldsA0 = (unsigned)(uintptr_t)&As[0][0][0];
    const unsigned ldsA1 = (unsigned)(uintptr_t)&As[1][0][0];
    const unsigned ldsB0 = (unsigned)(uintptr_t)&Bs[0][0][0];
    const unsigned ldsB1 = (unsigned)(uintptr_t)&Bs[1][0][0];

    v8f acc[2][2];
    const v8f vz = {0.f,0.f,0.f,0.f,0.f,0.f,0.f,0.f};
    acc[0][0] = vz; acc[0][1] = vz; acc[1][0] = vz; acc[1][1] = vz;

    const int ar0 = wm * 32 + ln, ar1 = ar0 + 16;
    const int br0 = wn * 32 + ln, br1 = br0 + 16;

    stage_tiles(gA, gB, ldsA0, ldsB0, tid, 0);
    const int NK = C_DIM / KSTEP;                 // 16
    for (int kk = 0; kk < NK; ++kk) {
        const int cur = kk & 1;
        if (kk + 1 < NK) {
            stage_tiles(gA, gB, cur ? ldsA0 : ldsA1, cur ? ldsB0 : ldsB1,
                        tid, (kk + 1) * KSTEP);
            WAIT_ASYNC(6);                        // own loads for `cur` retired
        } else {
            WAIT_ASYNC(0);
        }
        __syncthreads();                          // all waves' `cur` tile visible

        const __bf16* Ab = &As[cur][0][0];
        const __bf16* Bb = &Bs[cur][0][0];
#pragma unroll
        for (int ks = 0; ks < KSTEP; ks += 32) {
            v16bf a0 = a_frag(Ab + ar0 * LDSROW, ks, hi);
            v16bf a1 = a_frag(Ab + ar1 * LDSROW, ks, hi);
            v16bf b0 = b_frag(Bb + br0 * LDSROW, ks, hi);
            v16bf b1 = b_frag(Bb + br1 * LDSROW, ks, hi);
            acc[0][0] = wmma_bf16(a0, b0, acc[0][0]);
            acc[0][1] = wmma_bf16(a0, b1, acc[0][1]);
            acc[1][0] = wmma_bf16(a1, b0, acc[1][0]);
            acc[1][1] = wmma_bf16(a1, b1, acc[1][1]);
        }
        __syncthreads();                          // done reading before overwrite
    }

    const int R0 = Rblk + wm * 32;
    const int C0 = Cblk + wn * 32;
#pragma unroll
    for (int mt = 0; mt < 2; ++mt) {
#pragma unroll
        for (int nt = 0; nt < 2; ++nt) {
            const int c  = C0 + nt * 16 + ln;          // output column (= h*64+d)
            const float bc = bias[c];
            const int d  = c & 63;
            const float invf = __expf(-(float)(d & ~1) * (LN10000 / 64.0f));
            const bool even = (d & 1) == 0;
            const int h_ = c >> 6;
#pragma unroll
            for (int j = 0; j < 8; ++j) {
                const int r  = R0 + mt * 16 + j + hi * 8;  // global row = b*T + t
                const int t  = r & (T_SEQ - 1);
                const int b_ = r >> 11;
                float val = acc[mt][nt][j] + bc;
                float res = val;
                if (z < 2) {  // RoPE on q and k
                    float ang = (float)t * invf;
                    float sn, cs;
                    __sincosf(ang, &sn, &cs);
                    float other = __shfl_xor(val, 1, 32);  // partner column d^1
                    res = even ? (val * cs - other * sn) : (other * sn + val * cs);
                }
                const int bh = b_ * N_HEAD + h_;
                const int bhtd = bh * (T_SEQ * HEAD_DIM) + t * HEAD_DIM + d;
                if (z == 0) {
                    qws[bhtd] = (__bf16)res;
                } else if (z == 1) {
                    kout[bhtd] = res;
                    kws[bhtd]  = (__bf16)res;
                } else {
                    vout[bhtd] = res;
                    vtws[bh * (HEAD_DIM * T_SEQ) + d * T_SEQ + t] = (__bf16)res;
                }
            }
        }
    }
}

// ---- wave-level causal flash attention ------------------------------------
__global__ __launch_bounds__(256)
void flash_attn(const __bf16* __restrict__ qws, const __bf16* __restrict__ kws,
                const __bf16* __restrict__ vtws, __bf16* __restrict__ yws) {
    __shared__ __align__(16) __bf16 plds[8][16][40];   // 32 cols + 8 pad

    const int lane = threadIdx.x & 31;
    const int wvid = threadIdx.x >> 5;
    const int gw = blockIdx.x * 8 + wvid;     // 0..8191
    const int qt = gw & 127;
    const int bh = gw >> 7;                   // 0..63
    const int h_ = bh & (N_HEAD - 1), b_ = bh >> 4;
    const int q0 = qt * 16;
    const int ln = lane & 15, hi = lane >> 4;

    const __bf16* Qrow = qws + (bh * T_SEQ + q0 + ln) * HEAD_DIM;
    const v16bf qa0 = a_frag(Qrow, 0, hi);
    const v16bf qa1 = a_frag(Qrow, 32, hi);

    const v8f vz = {0.f,0.f,0.f,0.f,0.f,0.f,0.f,0.f};
    v8f O[4];
    O[0] = vz; O[1] = vz; O[2] = vz; O[3] = vz;
    float mi[8], li[8];
#pragma unroll
    for (int j = 0; j < 8; ++j) { mi[j] = -3.0e38f; li[j] = 0.f; }

    const float scale = 0.125f;               // 1/sqrt(64)
    const int nb = (q0 + 47) >> 5;            // causal key blocks

    for (int kb32 = 0; kb32 < nb; ++kb32) {
        const int kb = kb32 * 32;
        v8f S[2];
        S[0] = vz; S[1] = vz;
#pragma unroll
        for (int f = 0; f < 2; ++f) {
            const __bf16* Krow = kws + (bh * T_SEQ + kb + f * 16 + ln) * HEAD_DIM;
            v16bf kf0 = b_frag(Krow, 0, hi);
            v16bf kf1 = b_frag(Krow, 32, hi);
            S[f] = wmma_bf16(qa0, kf0, S[f]);
            S[f] = wmma_bf16(qa1, kf1, S[f]);
        }
#pragma unroll
        for (int j = 0; j < 8; ++j) {
            const int t = q0 + j + hi * 8;
            float s0 = S[0][j] * scale;
            float s1 = S[1][j] * scale;
            if (kb + ln > t)      s0 = -3.0e38f;
            if (kb + 16 + ln > t) s1 = -3.0e38f;
            float mx = fmaxf(s0, s1);
            mx = fmaxf(mx, __shfl_xor(mx, 1, 32));
            mx = fmaxf(mx, __shfl_xor(mx, 2, 32));
            mx = fmaxf(mx, __shfl_xor(mx, 4, 32));
            mx = fmaxf(mx, __shfl_xor(mx, 8, 32));
            const float mnew  = fmaxf(mi[j], mx);
            const float alpha = __expf(mi[j] - mnew);
            const float p0 = __expf(s0 - mnew);
            const float p1 = __expf(s1 - mnew);
            float rs = p0 + p1;
            rs += __shfl_xor(rs, 1, 32);
            rs += __shfl_xor(rs, 2, 32);
            rs += __shfl_xor(rs, 4, 32);
            rs += __shfl_xor(rs, 8, 32);
            li[j] = li[j] * alpha + rs;
            mi[j] = mnew;
            plds[wvid][j + hi * 8][ln]      = (__bf16)p0;
            plds[wvid][j + hi * 8][16 + ln] = (__bf16)p1;
            O[0][j] *= alpha; O[1][j] *= alpha; O[2][j] *= alpha; O[3][j] *= alpha;
        }
        BF16x16 pu;
        pu.h[0] = *(const v8bf*)&plds[wvid][ln][hi * 16];
        pu.h[1] = *(const v8bf*)&plds[wvid][ln][hi * 16 + 8];
        const v16bf pa = pu.v;
#pragma unroll
        for (int dt = 0; dt < 4; ++dt) {
            const __bf16* Vcol = vtws + (bh * HEAD_DIM + dt * 16 + ln) * T_SEQ + kb + hi * 16;
            BF16x16 vb;
            vb.h[0] = *(const v8bf*)(Vcol);
            vb.h[1] = *(const v8bf*)(Vcol + 8);
            O[dt] = wmma_bf16(pa, vb.v, O[dt]);
        }
    }

#pragma unroll
    for (int j = 0; j < 8; ++j) li[j] = 1.0f / li[j];
#pragma unroll
    for (int dt = 0; dt < 4; ++dt) {
#pragma unroll
        for (int j = 0; j < 8; ++j) {
            const int t = q0 + j + hi * 8;
            yws[(b_ * T_SEQ + t) * C_DIM + h_ * HEAD_DIM + dt * 16 + ln] =
                (__bf16)(O[dt][j] * li[j]);
        }
    }
}

// ---- output projection: out = y @ Wp^T + bp (LDS-staged) ------------------
__global__ __launch_bounds__(256)
void proj_gemm(const __bf16* __restrict__ yb, const __bf16* __restrict__ wp,
               const float* __restrict__ bp, float* __restrict__ out) {
    __shared__ __align__(16) __bf16 As[2][128][LDSROW];
    __shared__ __align__(16) __bf16 Bs[2][64][LDSROW];

    const int tid  = threadIdx.x;
    const int lane = tid & 31;
    const int wid  = tid >> 5;
    const int wm = wid & 3, wn = wid >> 2;
    const int Rblk = blockIdx.x * 128;
    const int Cblk = blockIdx.y * 64;
    const int ln = lane & 15, hi = lane >> 4;

    const __bf16* gA = yb + Rblk * C_DIM;
    const __bf16* gB = wp + Cblk * C_DIM;
    const unsigned ldsA0 = (unsigned)(uintptr_t)&As[0][0][0];
    const unsigned ldsA1 = (unsigned)(uintptr_t)&As[1][0][0];
    const unsigned ldsB0 = (unsigned)(uintptr_t)&Bs[0][0][0];
    const unsigned ldsB1 = (unsigned)(uintptr_t)&Bs[1][0][0];

    v8f acc[2][2];
    const v8f vz = {0.f,0.f,0.f,0.f,0.f,0.f,0.f,0.f};
    acc[0][0] = vz; acc[0][1] = vz; acc[1][0] = vz; acc[1][1] = vz;

    const int ar0 = wm * 32 + ln, ar1 = ar0 + 16;
    const int br0 = wn * 32 + ln, br1 = br0 + 16;

    stage_tiles(gA, gB, ldsA0, ldsB0, tid, 0);
    const int NK = C_DIM / KSTEP;
    for (int kk = 0; kk < NK; ++kk) {
        const int cur = kk & 1;
        if (kk + 1 < NK) {
            stage_tiles(gA, gB, cur ? ldsA0 : ldsA1, cur ? ldsB0 : ldsB1,
                        tid, (kk + 1) * KSTEP);
            WAIT_ASYNC(6);
        } else {
            WAIT_ASYNC(0);
        }
        __syncthreads();

        const __bf16* Ab = &As[cur][0][0];
        const __bf16* Bb = &Bs[cur][0][0];
#pragma unroll
        for (int ks = 0; ks < KSTEP; ks += 32) {
            v16bf a0 = a_frag(Ab + ar0 * LDSROW, ks, hi);
            v16bf a1 = a_frag(Ab + ar1 * LDSROW, ks, hi);
            v16bf b0 = b_frag(Bb + br0 * LDSROW, ks, hi);
            v16bf b1 = b_frag(Bb + br1 * LDSROW, ks, hi);
            acc[0][0] = wmma_bf16(a0, b0, acc[0][0]);
            acc[0][1] = wmma_bf16(a0, b1, acc[0][1]);
            acc[1][0] = wmma_bf16(a1, b0, acc[1][0]);
            acc[1][1] = wmma_bf16(a1, b1, acc[1][1]);
        }
        __syncthreads();
    }

    const int R0 = Rblk + wm * 32;
    const int C0 = Cblk + wn * 32;
#pragma unroll
    for (int mt = 0; mt < 2; ++mt) {
#pragma unroll
        for (int nt = 0; nt < 2; ++nt) {
            const int c = C0 + nt * 16 + ln;
            const float bc = bp[c];
#pragma unroll
            for (int j = 0; j < 8; ++j) {
                const int r = R0 + mt * 16 + j + hi * 8;
                out[r * C_DIM + c] = acc[mt][nt][j] + bc;
            }
        }
    }
}

// ---- launcher --------------------------------------------------------------
extern "C" void kernel_launch(void* const* d_in, const int* in_sizes, int n_in,
                              void* d_out, int out_size, void* d_ws, size_t ws_size,
                              hipStream_t stream) {
    const float* x  = (const float*)d_in[0];
    const float* Wq = (const float*)d_in[1];
    const float* bq = (const float*)d_in[2];
    const float* Wk = (const float*)d_in[3];
    const float* bk = (const float*)d_in[4];
    const float* Wv = (const float*)d_in[5];
    const float* bv = (const float*)d_in[6];
    const float* Wp = (const float*)d_in[7];
    const float* bp = (const float*)d_in[8];

    float* out  = (float*)d_out;                    // (B,T,C)
    float* kout = out + 8388608;                    // (B,H,T,D)
    float* vout = out + 16777216;                   // (B,H,T,D)

    char* ws = (char*)d_ws;
    const size_t XEL = 8388608, WEL = 1048576;
    __bf16* xb  = (__bf16*)(ws);                         // 16 MB (reused as y)
    __bf16* wqb = (__bf16*)(ws + 16777216);
    __bf16* wkb = (__bf16*)(ws + 18874368);
    __bf16* wvb = (__bf16*)(ws + 20971520);
    __bf16* wpb = (__bf16*)(ws + 23068672);
    __bf16* qb  = (__bf16*)(ws + 25165824);              // (B,H,T,D) bf16
    __bf16* kb  = (__bf16*)(ws + 41943040);              // (B,H,T,D) bf16
    __bf16* vtb = (__bf16*)(ws + 58720256);              // (B,H,D,T) bf16
    __bf16* yb  = xb;                                    // (B,T,H,D) bf16, aliases xb

    cvt_f32_bf16<<<dim3(XEL / 4 / 256), 256, 0, stream>>>(x,  xb,  (int)XEL);
    cvt_f32_bf16<<<dim3(WEL / 4 / 256), 256, 0, stream>>>(Wq, wqb, (int)WEL);
    cvt_f32_bf16<<<dim3(WEL / 4 / 256), 256, 0, stream>>>(Wk, wkb, (int)WEL);
    cvt_f32_bf16<<<dim3(WEL / 4 / 256), 256, 0, stream>>>(Wv, wvb, (int)WEL);
    cvt_f32_bf16<<<dim3(WEL / 4 / 256), 256, 0, stream>>>(Wp, wpb, (int)WEL);

    qkv_gemm_rope<<<dim3(ROWS / 128, C_DIM / 64, 3), 256, 0, stream>>>(
        xb, wqb, wkb, wvb, bq, bk, bv, qb, kb, vtb, kout, vout);

    flash_attn<<<dim3(1024), 256, 0, stream>>>(qb, kb, vtb, yb);

    proj_gemm<<<dim3(ROWS / 128, C_DIM / 64), 256, 0, stream>>>(yb, wpb, bp, out);
}
